// MultiHeadAttention_90409061580949
// MI455X (gfx1250) — matrix operand order repaired and checked by reading, not compiled, not run on hardware
//
#include <hip/hip_runtime.h>
#include <hip/hip_bf16.h>

// Problem constants (fixed by the reference)
constexpr int kE = 1024;   // embed dim
constexpr int kS = 2048;   // sequence
constexpr int kB = 2;      // batch
constexpr int kH = 16;     // heads
constexpr int kD = 64;     // head dim

typedef __attribute__((ext_vector_type(16))) __bf16 v16bf;
typedef __attribute__((ext_vector_type(8)))  __bf16 v8bf;
typedef __attribute__((ext_vector_type(8)))  float  v8f;
typedef __attribute__((ext_vector_type(4)))  unsigned int u32x4;
typedef __attribute__((ext_vector_type(8)))  int i32x8;
typedef __attribute__((ext_vector_type(4)))  int i32x4;

static __device__ __forceinline__ v8f wmma_bf16(v16bf a, v16bf b, v8f c) {
  // D = A(16x32 bf16) x B(32x16 bf16) + C(16x16 f32)
  return __builtin_amdgcn_wmma_f32_16x16x32_bf16(false, a, false, b, (short)0, c,
                                                 false, false);
}

static __device__ __forceinline__ v16bf cat8(v8bf lo, v8bf hi) {
  return __builtin_shufflevector(lo, hi, 0, 1, 2, 3, 4, 5, 6, 7, 8, 9, 10, 11,
                                 12, 13, 14, 15);
}

static __device__ __forceinline__ v8f zero8() {
  v8f z = {0.f, 0.f, 0.f, 0.f, 0.f, 0.f, 0.f, 0.f};
  return z;
}

// ---------------------------------------------------------------------------
// TDM: DMA a 128(rows) x 64(bf16) tile from global into LDS with an 80-element
// padded row stride (pad 8 DWORDs after every 32 DWORDs = one 128B row).
// D# layout per CDNA5 ISA ch.10.8 / 8.x; 6-arg (clang-23) builtin form.
// Issue from one wave; EXEC is ignored by TDM. Complete with s_wait_tensorcnt.
// ---------------------------------------------------------------------------
static __device__ __forceinline__ void tdm_load_k_tile(const __bf16* src,
                                                       unsigned int lds_off) {
  unsigned long long ga = (unsigned long long)(uintptr_t)src;
  u32x4 g0;
  g0[0] = 1u;                                    // count=1, user descriptor
  g0[1] = lds_off;                               // lds_addr (bytes)
  g0[2] = (unsigned int)ga;                      // global_addr[31:0]
  g0[3] = (unsigned int)((ga >> 32) & 0x1FFFFFFu) | (2u << 30);  // addr[56:32], type=2
  i32x8 g1;
  g1[0] = (1 << 16)        // data_size = 2 bytes
        | (1 << 20)        // pad_enable
        | (4 << 22)        // pad_interval: 32 DWORDs (one 64-elem row)
        | (7 << 25);       // pad_amount: 8 DWORDs (16 elems -> stride 80)
  g1[1] = (64 << 16);      // tensor_dim0 = 64 (bits 79:48, low 16 here)
  g1[2] = (int)(2048u << 16);  // tensor_dim0 hi=0 | tensor_dim1 lo16 = 2048
  g1[3] = (64 << 16);      // tensor_dim1 hi=0 | tile_dim0 = 64
  g1[4] = 128;             // tile_dim1 = 128 | tile_dim2 = 0
  g1[5] = 64;              // tensor_dim0_stride = 64 (low 32)
  g1[6] = 0;               // stride0 hi | stride1 lo
  g1[7] = 0;               // stride1 hi
  i32x4 z4 = {0, 0, 0, 0};
  i32x8 z8 = {0, 0, 0, 0, 0, 0, 0, 0};
  __builtin_amdgcn_tensor_load_to_lds(g0, g1, z4, z4, z8, 0);
}

// ---------------------------------------------------------------------------
// Linear layer: C[m,n] = (sum_k A[m,k] * W[n,k] + bias[n]) * oscale
//   A: [4096 x 1024] (f32 if !ABF, bf16 if ABF); W: [1024 x 1024] f32
// MODE 0: write bf16 to attention layout [b][h][s][d]
// MODE 1: write f32 row-major [m][n]
// 256 threads / 8 waves; block tile 128x64; wave tile 32x32 (2x2 WMMA); K=32.
// ---------------------------------------------------------------------------
template <int MODE, bool ABF>
__global__ __launch_bounds__(256) void linear_wmma(
    const float* __restrict__ Af, const __bf16* __restrict__ Ab,
    const float* __restrict__ W, const float* __restrict__ bias,
    __bf16* __restrict__ outb, float* __restrict__ outf, float oscale) {
  __shared__ __bf16 As[128][48];  // stride 48: v8bf reads 16B aligned
  __shared__ __bf16 Bs[64][48];   // stride 48: v16bf reads 32B aligned

  const int tid = threadIdx.x;
  const int lane = tid & 31, w = tid >> 5;
  const int wm = w & 3, wn = w >> 2;
  const int half = lane >> 4, l16 = lane & 15;
  const int mblock = blockIdx.y * 128;
  const int nblock = blockIdx.x * 64;

  v8f acc[2][2];
#pragma unroll
  for (int i = 0; i < 2; ++i)
#pragma unroll
    for (int j = 0; j < 2; ++j) acc[i][j] = zero8();

  for (int k0 = 0; k0 < kE; k0 += 32) {
    // A tile (128x32), vectorized: 8 chunks of 4 per row
    for (int e = tid; e < 128 * 8; e += 256) {
      const int r = e >> 3, c4 = (e & 7) * 4;
      union { __bf16 h[4]; uint2 u; } pk;
      if (ABF) {
        pk.u = *(const uint2*)&Ab[(size_t)(mblock + r) * kE + k0 + c4];
      } else {
        float4 fv = *(const float4*)&Af[(size_t)(mblock + r) * kE + k0 + c4];
        pk.h[0] = (__bf16)fv.x; pk.h[1] = (__bf16)fv.y;
        pk.h[2] = (__bf16)fv.z; pk.h[3] = (__bf16)fv.w;
      }
      *(uint2*)&As[r][c4] = pk.u;
    }
    // W tile (64x32)
    for (int e = tid; e < 64 * 8; e += 256) {
      const int r = e >> 3, c4 = (e & 7) * 4;
      float4 fv = *(const float4*)&W[(size_t)(nblock + r) * kE + k0 + c4];
      union { __bf16 h[4]; uint2 u; } pk;
      pk.h[0] = (__bf16)fv.x; pk.h[1] = (__bf16)fv.y;
      pk.h[2] = (__bf16)fv.z; pk.h[3] = (__bf16)fv.w;
      *(uint2*)&Bs[r][c4] = pk.u;
    }
    __syncthreads();

    v16bf bfrag[2];
#pragma unroll
    for (int j = 0; j < 2; ++j)
      bfrag[j] = *(const v16bf*)&Bs[wn * 32 + j * 16 + l16][half * 16];

#pragma unroll
    for (int i = 0; i < 2; ++i) {
      const __bf16* ar = &As[wm * 32 + i * 16 + l16][0];
      v8bf lo = *(const v8bf*)(ar + half * 16);
      v8bf hi = *(const v8bf*)(ar + half * 16 + 8);
      v16bf afrag = cat8(lo, hi);
#pragma unroll
      for (int j = 0; j < 2; ++j) acc[i][j] = wmma_bf16(afrag, bfrag[j], acc[i][j]);
    }
    __syncthreads();
  }

  // epilogue: C layout lane -> (m = r + 8*half, n = l16)
#pragma unroll
  for (int i = 0; i < 2; ++i) {
#pragma unroll
    for (int j = 0; j < 2; ++j) {
      const int col = nblock + wn * 32 + j * 16 + l16;
      const float bv = bias[col];
#pragma unroll
      for (int r = 0; r < 8; ++r) {
        const int row = mblock + wm * 32 + i * 16 + r + 8 * half;
        const float vv = (acc[i][j][r] + bv) * oscale;
        if (MODE == 0) {
          const int bb = row >> 11, s = row & (kS - 1);
          const int hh = col >> 6, d = col & (kD - 1);
          outb[(((size_t)bb * kH + hh) * kS + s) * kD + d] = (__bf16)vv;
        } else {
          outf[(size_t)row * kE + col] = vv;
        }
      }
    }
  }
}

// ---------------------------------------------------------------------------
// Fused causal attention, one block = (b, h, 128-query tile), 8 waves x 16
// query rows. Q is pre-scaled by 1/sqrt(d) in the projection.
//   pass 1: branchless online row max/sum over key tiles (K tiles via TDM)
//   pass 2: recompute scores, write normalized probs (attn streamed once),
//           accumulate O = P @ V via WMMA through a per-wave LDS P buffer.
// ---------------------------------------------------------------------------
__global__ __launch_bounds__(256) void attention_wmma(
    const __bf16* __restrict__ q, const __bf16* __restrict__ k,
    const __bf16* __restrict__ v, float* __restrict__ attn,
    __bf16* __restrict__ ctx) {
  __shared__ __bf16 Ks[128][80];    // [key][d], TDM-filled, 20 KB
  __shared__ __bf16 Vt[64][144];    // [d][key], 18 KB
  __shared__ __bf16 Ps[8][16][40];  // per-wave P chunk, 10 KB

  const int qt = blockIdx.x;
  const int h = blockIdx.y, b = blockIdx.z;
  const int bh = b * kH + h;
  const int tid = threadIdx.x;
  const int lane = tid & 31, w = tid >> 5;
  const int half = lane >> 4, l16 = lane & 15;
  const int qrow0 = qt * 128 + w * 16;

  const __bf16* qb = q + (size_t)bh * kS * kD;
  const __bf16* kbp = k + (size_t)bh * kS * kD;
  const __bf16* vbp = v + (size_t)bh * kS * kD;
  float* attnBH = attn + (size_t)bh * kS * kS;
  const unsigned int ldsK = (unsigned int)(uintptr_t)(void*)&Ks[0][0];

  // Q fragments (16 rows x 64 d), held in registers for the whole block
  v16bf qa[2];
#pragma unroll
  for (int c = 0; c < 2; ++c) {
    const __bf16* p = qb + (size_t)(qrow0 + l16) * kD + c * 32;
    v8bf lo = *(const v8bf*)(p + half * 16);
    v8bf hi = *(const v8bf*)(p + half * 16 + 8);
    qa[c] = cat8(lo, hi);
  }

  float mrow[8], lrow[8];
#pragma unroll
  for (int r = 0; r < 8; ++r) {
    mrow[r] = -1e30f;
    lrow[r] = 0.f;
  }

  // -------- pass 1: row max / sum (branchless per-element updates) --------
  for (int kt = 0; kt <= qt; ++kt) {
    if (w == 0) {
      tdm_load_k_tile(kbp + (size_t)kt * 128 * kD, ldsK);
      __builtin_amdgcn_s_wait_tensorcnt(0);
    }
    __syncthreads();
    for (int t16 = 0; t16 < 8; ++t16) {
      if (kt * 128 + t16 * 16 > qrow0 + 15) continue;  // fully-masked subtile
      v8f s = zero8();
#pragma unroll
      for (int c = 0; c < 2; ++c) {
        v16bf bf = *(const v16bf*)&Ks[t16 * 16 + l16][c * 32 + half * 16];
        s = wmma_bf16(qa[c], bf, s);
      }
      const int kidx = kt * 128 + t16 * 16 + l16;
#pragma unroll
      for (int r = 0; r < 8; ++r) {
        const int qidx = qrow0 + r + 8 * half;
        const bool ok = kidx <= qidx;
        const float msk = ok ? 1.0f : 0.0f;          // v_cndmask, no branch
        const float svm = ok ? s[r] : -1e30f;        // v_cndmask, no branch
        const float mo = mrow[r];
        const float mn = fmaxf(mo, svm);
        lrow[r] = lrow[r] * __expf(mo - mn) + msk * __expf(svm - mn);
        mrow[r] = mn;
      }
    }
    __syncthreads();
  }

  // cross-lane reduce within the 16-lane groups that share rows
#pragma unroll
  for (int off = 1; off < 16; off <<= 1) {
#pragma unroll
    for (int r = 0; r < 8; ++r) {
      const float m2 = __shfl_xor(mrow[r], off);
      const float l2 = __shfl_xor(lrow[r], off);
      const float mn = fmaxf(mrow[r], m2);
      lrow[r] = lrow[r] * __expf(mrow[r] - mn) + l2 * __expf(m2 - mn);
      mrow[r] = mn;
    }
  }
  float rl[8];
#pragma unroll
  for (int r = 0; r < 8; ++r) rl[r] = __builtin_amdgcn_rcpf(lrow[r]);

  // -------- pass 2: recompute, emit attn, accumulate O = P @ V --------
  v8f oacc[4];
#pragma unroll
  for (int dt = 0; dt < 4; ++dt) oacc[dt] = zero8();

  for (int kt = 0; kt <= qt; ++kt) {
    if (w == 0) tdm_load_k_tile(kbp + (size_t)kt * 128 * kD, ldsK);
    // V tile, transposed into LDS (vectorized global reads)
    for (int e = tid; e < 128 * 16; e += 256) {
      const int kr = e >> 4, d4 = (e & 15) * 4;
      union { uint2 u; __bf16 h[4]; } pk;
      pk.u = *(const uint2*)&vbp[(size_t)(kt * 128 + kr) * kD + d4];
      Vt[d4 + 0][kr] = pk.h[0];
      Vt[d4 + 1][kr] = pk.h[1];
      Vt[d4 + 2][kr] = pk.h[2];
      Vt[d4 + 3][kr] = pk.h[3];
    }
    if (w == 0) __builtin_amdgcn_s_wait_tensorcnt(0);
    __syncthreads();

    for (int c = 0; c < 4; ++c) {  // 32-key chunks
#pragma unroll
      for (int sub = 0; sub < 2; ++sub) {
        const int t16 = c * 2 + sub;
        const bool live = (kt * 128 + t16 * 16) <= (qrow0 + 15);  // wave-uniform
        v8f s = zero8();
        if (live) {
#pragma unroll
          for (int cc = 0; cc < 2; ++cc) {
            v16bf bf = *(const v16bf*)&Ks[t16 * 16 + l16][cc * 32 + half * 16];
            s = wmma_bf16(qa[cc], bf, s);
          }
        }
        const int kidx = kt * 128 + t16 * 16 + l16;
#pragma unroll
        for (int r = 0; r < 8; ++r) {
          const int qidx = qrow0 + r + 8 * half;
          // exp(-1e30 - m) == 0 exactly, so masked elements yield p == 0
          const float svm = (kidx <= qidx) ? s[r] : -1e30f;
          const float p = __expf(svm - mrow[r]) * rl[r];
          attnBH[(size_t)qidx * kS + kidx] = p;             // streamed once
          Ps[w][r + 8 * half][sub * 16 + l16] = (__bf16)p;  // C->A relayout
        }
      }
      // P A-fragment for this 32-key chunk (wave-private LDS, in-order DS)
      v8bf lo = *(const v8bf*)&Ps[w][l16][half * 16];
      v8bf hi = *(const v8bf*)&Ps[w][l16][half * 16 + 8];
      v16bf pa = cat8(lo, hi);
#pragma unroll
      for (int dt = 0; dt < 4; ++dt) {
        v16bf vf = *(const v16bf*)&Vt[dt * 16 + l16][c * 32 + half * 16];
        oacc[dt] = wmma_bf16(pa, vf, oacc[dt]);
      }
    }
    __syncthreads();
  }

  // context (bf16, [b][s][E]) for the output projection
#pragma unroll
  for (int dt = 0; dt < 4; ++dt) {
#pragma unroll
    for (int r = 0; r < 8; ++r) {
      const int qidx = qrow0 + r + 8 * half;
      const int d = dt * 16 + l16;
      ctx[((size_t)b * kS + qidx) * kE + h * kD + d] = (__bf16)oacc[dt][r];
    }
  }

  // zero-fill strictly masked key tiles (softmax of -inf rows is exact 0)
  const int zcols = kS - (qt + 1) * 128;
  if (zcols > 0) {
    const int c4h = zcols >> 3;  // half-row in float4 units (zcols mult of 128)
    const int row = tid >> 1;
    float* dst = attnBH + (size_t)(qt * 128 + row) * kS + (qt + 1) * 128 +
                 (tid & 1) * (size_t)c4h * 4;
    float4 z = make_float4(0.f, 0.f, 0.f, 0.f);
    for (int cc = 0; cc < c4h; ++cc) *(float4*)(dst + cc * 4) = z;
  }
}

// ---------------------------------------------------------------------------
extern "C" void kernel_launch(void* const* d_in, const int* in_sizes, int n_in,
                              void* d_out, int out_size, void* d_ws,
                              size_t ws_size, hipStream_t stream) {
  const float* query = (const float*)d_in[0];
  const float* key = (const float*)d_in[1];
  const float* value = (const float*)d_in[2];
  const float* Wq = (const float*)d_in[3];
  const float* bq = (const float*)d_in[4];
  const float* Wk = (const float*)d_in[5];
  const float* bk = (const float*)d_in[6];
  const float* Wv = (const float*)d_in[7];
  const float* bv = (const float*)d_in[8];
  const float* Wo = (const float*)d_in[9];
  const float* bo = (const float*)d_in[10];

  float* outp = (float*)d_out;                // [B,S,E] f32
  float* attn = outp + (size_t)kB * kS * kE;  // [B,H,S,S] f32

  const size_t act = (size_t)kB * kS * kE;
  __bf16* qws = (__bf16*)d_ws;  // bf16 q/k/v in [b][h][s][d]
  __bf16* kws = qws + act;
  __bf16* vws = kws + act;
  __bf16* cws = vws + act;      // bf16 ctx [b][s][E]

  dim3 blk(256);
  dim3 g1(kE / 64, (kB * kS) / 128);
  // Q pre-scaled by 1/sqrt(head_dim) so softmax passes skip the multiply
  linear_wmma<0, false><<<g1, blk, 0, stream>>>(query, nullptr, Wq, bq, qws,
                                                nullptr, 0.125f);
  linear_wmma<0, false><<<g1, blk, 0, stream>>>(key, nullptr, Wk, bk, kws,
                                                nullptr, 1.0f);
  linear_wmma<0, false><<<g1, blk, 0, stream>>>(value, nullptr, Wv, bv, vws,
                                                nullptr, 1.0f);

  dim3 g2(kS / 128, kH, kB);
  attention_wmma<<<g2, blk, 0, stream>>>(qws, kws, vws, attn, cws);

  linear_wmma<1, true><<<g1, blk, 0, stream>>>(nullptr, cws, Wo, bo, nullptr,
                                               outp, 1.0f);
}
